// CausalSelfAttention_65360812310831
// MI455X (gfx1250) — matrix operand order repaired and checked by reading, not compiled, not run on hardware
//
#include <hip/hip_runtime.h>
#include <hip/hip_bf16.h>

// ---------------- constants (shapes fixed by the reference) ----------------
#define BB     2
#define TT     1024
#define PAST   1024
#define TOTAL  2048
#define HH     16
#define HKV    8
#define DD     128
#define CC     2048          // H*D == 2*HKV*D
#define LN1E4  9.210340371976184f   // ln(10000)

typedef __attribute__((ext_vector_type(16))) __bf16   v16bf;
typedef __attribute__((ext_vector_type(8)))  float    v8f;
typedef __attribute__((ext_vector_type(4)))  unsigned v4u;
typedef __attribute__((ext_vector_type(8)))  unsigned v8u;

static __device__ __forceinline__ v8f v8f_zero() {
  v8f v = {0.f,0.f,0.f,0.f,0.f,0.f,0.f,0.f};
  return v;
}

// Build a 16xbf16 fragment (8 VGPRs) from two 16-byte chunks.
static __device__ __forceinline__ v16bf frag_from2(const __bf16* p0, const __bf16* p1) {
  v4u a = *reinterpret_cast<const v4u*>(p0);
  v4u b = *reinterpret_cast<const v4u*>(p1);
  v8u r = {a[0], a[1], a[2], a[3], b[0], b[1], b[2], b[3]};
  return __builtin_bit_cast(v16bf, r);
}

static __device__ __forceinline__ v8f wmma_bf16(v16bf a, v16bf b, v8f c) {
  // 8 args: (neg_a, A, neg_b, B, c_mod, C, reuse_a, reuse_b)
  return __builtin_amdgcn_wmma_f32_16x16x32_bf16(false, a, false, b, (short)0, c, false, false);
}

// Exchange data between lane halves (xor 16) via DS_SWIZZLE_B32.
// offset = xor_mask(0x10)<<10 | and_mask(0x1f) = 0x401F, group-of-32 mode.
static __device__ __forceinline__ float swapx16f(float x) {
  int y = __builtin_amdgcn_ds_swizzle(__builtin_bit_cast(int, x), 0x401F);
  return __builtin_bit_cast(float, y);
}
static __device__ __forceinline__ unsigned swapx16u(unsigned x) {
  return (unsigned)__builtin_amdgcn_ds_swizzle((int)x, 0x401F);
}

static __device__ __forceinline__ unsigned pack_bf16(float a, float b) {
  unsigned short la = __builtin_bit_cast(unsigned short, (__bf16)a);
  unsigned short lb = __builtin_bit_cast(unsigned short, (__bf16)b);
  return (unsigned)la | ((unsigned)lb << 16);
}

// CDNA5 async global->LDS copy: 16 bytes per lane, tracked by ASYNCcnt.
// vdst = per-lane LDS byte address, vaddr = per-lane 64-bit global address.
static __device__ __forceinline__ void async_load_b128(unsigned lds_off, const void* gaddr) {
  asm volatile("global_load_async_to_lds_b128 %0, %1, off"
               :: "v"(lds_off), "v"(gaddr) : "memory");
}
static __device__ __forceinline__ void wait_asynccnt0() {
  asm volatile("s_wait_asynccnt 0x0" ::: "memory");
}

// ---------------- kernel 0: fp32 -> bf16 convert (4 elems/thread) ----------------
__global__ void __launch_bounds__(256) cvt_f32_bf16(const float* __restrict__ s,
                                                    __bf16* __restrict__ d, int n) {
  int i = (blockIdx.x * blockDim.x + threadIdx.x) * 4;
  if (i + 3 < n) {
    float4 v = *reinterpret_cast<const float4*>(s + i);
    unsigned lo = pack_bf16(v.x, v.y);
    unsigned hi = pack_bf16(v.z, v.w);
    *reinterpret_cast<unsigned*>(d + i)     = lo;
    *reinterpret_cast<unsigned*>(d + i + 2) = hi;
  }
}

// ---------------- kernel 1: C[M,N] (fp32) = A[M,K] * B[N,K]^T, bf16 operands ----------------
// Block tile 128(M) x 256(N), 8 waves of 64x64. K-slabs of 32 staged through LDS
// by async global->LDS loads, double buffered; WMMA consumes from LDS.
__global__ void __launch_bounds__(256) gemm_bf16_lds(const __bf16* __restrict__ A,
                                                     const __bf16* __restrict__ B,
                                                     float* __restrict__ C,
                                                     int M, int N, int K) {
  __shared__ __bf16 sA[2][128 * 32];   // 16 KB
  __shared__ __bf16 sB[2][256 * 32];   // 32 KB

  const int tid  = threadIdx.x;
  const int lane = tid & 31;
  const int w    = tid >> 5;           // 0..7
  const int mi   = w >> 2;             // 0..1  (64-row slab)
  const int ni   = w & 3;              // 0..3  (64-col slab)
  const int g = lane >> 4, r = lane & 15;

  const int bn = blockIdx.x & 7;       // N/256 = 8
  const int bm = blockIdx.x >> 3;      // M/128 = 16
  const __bf16* Abase = A + (size_t)bm * 128 * K;
  const __bf16* Bbase = B + (size_t)bn * 256 * K;

  const unsigned aoff0 = (unsigned)(uintptr_t)&sA[0][0];
  const unsigned aoff1 = (unsigned)(uintptr_t)&sA[1][0];
  const unsigned boff0 = (unsigned)(uintptr_t)&sB[0][0];
  const unsigned boff1 = (unsigned)(uintptr_t)&sB[1][0];

  // issue one K-slab (32 cols) of A(128 rows) + B(256 rows) into buffer `buf`
  auto issue_slab = [&](int kk, int buf) {
    unsigned ao = buf ? aoff1 : aoff0;
    unsigned bo = buf ? boff1 : boff0;
#pragma unroll
    for (int j = 0; j < 2; j++) {                 // A: 512 x 16B units
      int u = tid + 256 * j;
      int row = u >> 2, c = u & 3;
      async_load_b128(ao + (unsigned)(row * 64 + c * 16),
                      Abase + (size_t)row * K + kk + c * 8);
    }
#pragma unroll
    for (int j = 0; j < 4; j++) {                 // B: 1024 x 16B units
      int u = tid + 256 * j;
      int row = u >> 2, c = u & 3;
      async_load_b128(bo + (unsigned)(row * 64 + c * 16),
                      Bbase + (size_t)row * K + kk + c * 8);
    }
  };

  v8f acc[4][4];
#pragma unroll
  for (int i = 0; i < 4; i++)
#pragma unroll
    for (int j = 0; j < 4; j++) acc[i][j] = v8f_zero();

  issue_slab(0, 0);

  for (int k0 = 0; k0 < K; k0 += 32) {
    const int buf = (k0 >> 5) & 1;
    wait_asynccnt0();          // my async writes into `buf` are in LDS
    __syncthreads();           // everyone's writes visible; prev reads done
    if (k0 + 32 < K) issue_slab(k0 + 32, buf ^ 1);

    const __bf16* la = sA[buf];
    const __bf16* lb = sB[buf];
    v16bf af[4], bfr[4];
#pragma unroll
    for (int i = 0; i < 4; i++) {
      // A fragment: lanes0-15 rows, K chunks at 8g and 16+8g
      const __bf16* pa = la + (mi * 64 + 16 * i + r) * 32;
      af[i] = frag_from2(pa + 8 * g, pa + 16 + 8 * g);
      // B fragment: lane = column, K chunks at 8g and 16+8g
      const __bf16* pb = lb + (ni * 64 + 16 * i + r) * 32;
      bfr[i] = frag_from2(pb + 8 * g, pb + 16 + 8 * g);
    }
#pragma unroll
    for (int i = 0; i < 4; i++)
#pragma unroll
      for (int j = 0; j < 4; j++) acc[i][j] = wmma_bf16(af[i], bfr[j], acc[i][j]);
  }

#pragma unroll
  for (int i = 0; i < 4; i++) {
#pragma unroll
    for (int j = 0; j < 4; j++) {
      float* crow = C + (size_t)(bm * 128 + mi * 64 + 16 * i + 8 * g) * N
                      + bn * 256 + ni * 64 + 16 * j + r;
#pragma unroll
      for (int rr = 0; rr < 8; rr++) crow[(size_t)rr * N] = acc[i][j][rr];
    }
  }
}

// ---------------- kernel 2: RoPE on Q, relayout to [b,h,t,d] bf16 ----------------
__global__ void __launch_bounds__(256) rope_q(const float* __restrict__ Qraw,
                                              __bf16* __restrict__ Qb) {
  int idx = blockIdx.x * blockDim.x + threadIdx.x;   // B*T*H*D = 4M
  if (idx >= BB * TT * HH * DD) return;
  int d = idx & 127;
  int h = (idx >> 7) & 15;
  int t = (idx >> 11) & 1023;
  int b = idx >> 21;
  size_t row = (size_t)(b * TT + t) * CC;
  float v  = Qraw[row + h * DD + d];
  int   dp = (d < 64) ? d + 64 : d - 64;
  float pv = Qraw[row + h * DD + dp];
  float rot = (d < 64) ? -pv : pv;
  float inv = __expf(-(float)(d & 63) * (LN1E4 / 64.f));  // YaRN scale == 1 here
  float ang = (float)(PAST + t) * inv;
  float o = v * __cosf(ang) + rot * __sinf(ang);
  Qb[((size_t)(b * HH + h) * TT + t) * DD + d] = (__bf16)o;
}

// ---------------- kernel 3: gather past K/V from pools, RoPE+append new, build K and V^T ----------------
__global__ void __launch_bounds__(256) kv_scatter(const float* __restrict__ KVraw,
                                                  const float* __restrict__ Kpool,
                                                  const float* __restrict__ Vpool,
                                                  const int*   __restrict__ slot_map,
                                                  __bf16* __restrict__ Kg,
                                                  __bf16* __restrict__ VgT) {
  int idx = blockIdx.x * blockDim.x + threadIdx.x;   // B*HKV*TOTAL*D = 4M
  if (idx >= BB * HKV * TOTAL * DD) return;
  int d  = idx & 127;
  int s  = (idx >> 7) & 2047;
  int hk = (idx >> 18) & 7;
  int b  = idx >> 21;
  float kval, vval;
  if (s < PAST) {
    int slot = slot_map[b * TOTAL + s];
    size_t p = ((size_t)slot * HKV + hk) * DD + d;
    kval = Kpool[p];
    vval = Vpool[p];
  } else {
    int t = s - PAST;
    size_t row = (size_t)(b * TT + t) * CC;
    float k0 = KVraw[row + hk * DD + d];
    int   dp = (d < 64) ? d + 64 : d - 64;
    float kp = KVraw[row + hk * DD + dp];
    float rot = (d < 64) ? -kp : kp;
    float inv = __expf(-(float)(d & 63) * (LN1E4 / 64.f));
    float ang = (float)(PAST + t) * inv;
    kval = k0 * __cosf(ang) + rot * __sinf(ang);
    vval = KVraw[row + HKV * DD + hk * DD + d];
  }
  Kg [((size_t)(b * HKV + hk) * TOTAL + s) * DD + d]     = (__bf16)kval;
  VgT[((size_t)(b * HKV + hk) * DD + d) * TOTAL + s]     = (__bf16)vval;
}

// ---------------- kernel 4: flash attention (transposed form), 1 wave = 16 queries ----------------
__global__ void __launch_bounds__(256) attn_fa(const __bf16* __restrict__ Qb,
                                               const __bf16* __restrict__ Kg,
                                               const __bf16* __restrict__ VgT,
                                               __bf16* __restrict__ O) {
  const int lane = threadIdx.x & 31;
  const int wid  = blockIdx.x * (blockDim.x >> 5) + (threadIdx.x >> 5); // 0..2047
  const int tqi = wid & 63;
  const int h   = (wid >> 6) & 15;
  const int b   = wid >> 10;
  const int t0  = tqi * 16;
  const int kvh = h >> 1;                    // GQA: repeat(K,2,axis=1)
  const int g = lane >> 4, q = lane & 15;
  const int tq = t0 + q;

  const __bf16* Qp = Qb  + (size_t)(b * HH  + h)   * TT    * DD;
  const __bf16* Kp = Kg  + (size_t)(b * HKV + kvh) * TOTAL * DD;
  const __bf16* Vp = VgT + (size_t)(b * HKV + kvh) * DD    * TOTAL;

  // Q^T B-fragments: lane = query, K dim = d. 4 frags cover D=128.
  v16bf qf[4];
#pragma unroll
  for (int j = 0; j < 4; j++) {
    const __bf16* p = Qp + (size_t)tq * DD + 32 * j + 8 * g;
    qf[j] = frag_from2(p, p + 16);
  }

  v8f acc[8];                                 // O^T: 8 frags of 16(d) x 16(q)
#pragma unroll
  for (int i = 0; i < 8; i++) acc[i] = v8f_zero();
  float m = -1e30f, l = 0.f;
  const float scale = 0.08838834764831845f;   // 1/sqrt(128)
  const int send = PAST + t0 + 16;

  for (int s0 = 0; s0 < send; s0 += 32) {
    // scores S^T = K * Q^T : two 16-key tiles
    v8f c0 = v8f_zero(), c1 = v8f_zero();
#pragma unroll
    for (int j = 0; j < 4; j++) {
      const __bf16* k0 = Kp + (size_t)(s0 + q) * DD + 32 * j + 8 * g;
      const __bf16* k1 = Kp + (size_t)(s0 + 16 + q) * DD + 32 * j + 8 * g;
      c0 = wmma_bf16(frag_from2(k0, k0 + 16), qf[j], c0);
      c1 = wmma_bf16(frag_from2(k1, k1 + 16), qf[j], c1);
    }
    // scale + causal mask (S^T: lane = query, VGPR r -> key = base + r)
    const int kb0 = s0 + 8 * g, kb1 = kb0 + 16;
    const int lim = PAST + tq;
    float mx = -1e30f;
    if (s0 + 31 <= PAST + t0) {               // chunk fully unmasked
#pragma unroll
      for (int r = 0; r < 8; r++) {
        c0[r] *= scale; c1[r] *= scale;
        mx = fmaxf(mx, fmaxf(c0[r], c1[r]));
      }
    } else {
#pragma unroll
      for (int r = 0; r < 8; r++) {
        c0[r] = (kb0 + r <= lim) ? c0[r] * scale : -1e30f;
        c1[r] = (kb1 + r <= lim) ? c1[r] * scale : -1e30f;
        mx = fmaxf(mx, fmaxf(c0[r], c1[r]));
      }
    }
    mx = fmaxf(mx, swapx16f(mx));             // combine key-halves
    float mnew = fmaxf(m, mx);
    float corr = __expf(m - mnew);
    float sum = 0.f;
#pragma unroll
    for (int r = 0; r < 8; r++) {
      c0[r] = __expf(c0[r] - mnew);
      c1[r] = __expf(c1[r] - mnew);
      sum += c0[r] + c1[r];
    }
    sum += swapx16f(sum);
    l = l * corr + sum;
    m = mnew;
#pragma unroll
    for (int i = 0; i < 8; i++) acc[i] *= corr;

    // Build P^T B-fragment (32 keys x 16 queries) from the two score tiles:
    // pack pairs of key-rows into bf16x2, then cross-half swizzle + select.
    unsigned bw[8];
#pragma unroll
    for (int j = 0; j < 4; j++) {
      unsigned qj = pack_bf16(c0[2 * j], c0[2 * j + 1]);  // keys {2j,2j+1} (+8 in hi half)
      unsigned pj = pack_bf16(c1[2 * j], c1[2 * j + 1]);  // keys {16+2j,..} (+8 in hi half)
      unsigned sq = swapx16u(qj), sp = swapx16u(pj);
      bw[j]     = (g == 0) ? qj : sp;   // lanes0-15: keys 0-7  | lanes16-31: keys 16-23
      bw[4 + j] = (g == 0) ? sq : pj;   // lanes0-15: keys 8-15 | lanes16-31: keys 24-31
    }
    v8u braw = {bw[0], bw[1], bw[2], bw[3], bw[4], bw[5], bw[6], bw[7]};
    v16bf pf = __builtin_bit_cast(v16bf, braw);

    // O^T += V^T * P^T : 8 d-row fragments
#pragma unroll
    for (int i = 0; i < 8; i++) {
      const __bf16* vp = Vp + (size_t)(16 * i + q) * TOTAL + s0 + 16 * g;
      acc[i] = wmma_bf16(frag_from2(vp, vp + 8), pf, acc[i]);
    }
  }

  const float inv_l = 1.0f / l;
  // store O row-major [(b*T + tq)][h*D + d], bf16, paired 4B stores
  __bf16* orow = O + (size_t)(b * TT + tq) * CC + h * DD;
#pragma unroll
  for (int i = 0; i < 8; i++) {
    int dbase = 16 * i + 8 * g;
#pragma unroll
    for (int jj = 0; jj < 4; jj++) {
      unsigned pk = pack_bf16(acc[i][2 * jj] * inv_l, acc[i][2 * jj + 1] * inv_l);
      *reinterpret_cast<unsigned*>(orow + dbase + 2 * jj) = pk;
    }
  }
}

// ---------------- host-side launcher ----------------
extern "C" void kernel_launch(void* const* d_in, const int* in_sizes, int n_in,
                              void* d_out, int out_size, void* d_ws, size_t ws_size,
                              hipStream_t stream) {
  const float* x     = (const float*)d_in[0];
  const float* Wq    = (const float*)d_in[1];
  const float* Wkv   = (const float*)d_in[2];
  const float* Wo    = (const float*)d_in[3];
  const float* Kpool = (const float*)d_in[4];
  const float* Vpool = (const float*)d_in[5];
  const int*   slot  = (const int*)d_in[6];
  (void)in_sizes; (void)n_in; (void)out_size; (void)ws_size;

  char* ws = (char*)d_ws;
  const size_t MB = 1u << 20;
  __bf16* xb    = (__bf16*)(ws + 0 * MB);
  __bf16* Wqb   = (__bf16*)(ws + 8 * MB);
  __bf16* Wkvb  = (__bf16*)(ws + 16 * MB);
  __bf16* Wob   = (__bf16*)(ws + 24 * MB);
  float*  Qraw  = (float*) (ws + 32 * MB);
  float*  KVraw = (float*) (ws + 48 * MB);
  __bf16* Qb    = (__bf16*)(ws + 64 * MB);
  __bf16* Kg    = (__bf16*)(ws + 72 * MB);
  __bf16* VgT   = (__bf16*)(ws + 80 * MB);
  __bf16* Obf   = (__bf16*)(ws + 88 * MB);

  const int NE = CC * CC;                       // 4M elements per 2048x2048 tensor
  cvt_f32_bf16<<<NE / 1024, 256, 0, stream>>>(x,   xb,   NE);
  cvt_f32_bf16<<<NE / 1024, 256, 0, stream>>>(Wq,  Wqb,  NE);
  cvt_f32_bf16<<<NE / 1024, 256, 0, stream>>>(Wkv, Wkvb, NE);
  cvt_f32_bf16<<<NE / 1024, 256, 0, stream>>>(Wo,  Wob,  NE);

  // Q = x Wq^T ; KV = x Wkv^T   (M=2048, N=2048, K=2048; 16x8 = 128 blocks)
  gemm_bf16_lds<<<128, 256, 0, stream>>>(xb, Wqb,  Qraw,  BB * TT, CC, CC);
  gemm_bf16_lds<<<128, 256, 0, stream>>>(xb, Wkvb, KVraw, BB * TT, CC, CC);

  rope_q    <<<(BB * TT * HH * DD) / 256, 256, 0, stream>>>(Qraw, Qb);
  kv_scatter<<<(BB * HKV * TOTAL * DD) / 256, 256, 0, stream>>>(KVraw, Kpool, Vpool, slot, Kg, VgT);

  // 2048 query tiles -> 2048 waves -> 256 blocks x 8 waves
  attn_fa<<<256, 256, 0, stream>>>(Qb, Kg, VgT, Obf);

  // out = O Wo^T -> fp32 d_out
  gemm_bf16_lds<<<128, 256, 0, stream>>>(Obf, Wob, (float*)d_out, BB * TT, CC, CC);
}
